// GatedAttention_45724221833328
// MI455X (gfx1250) — matrix operand order repaired and checked by reading, not compiled, not run on hardware
//
#include <hip/hip_runtime.h>
#include <cstdint>

#define NUM_IN 1024
#define HIDDEN 32

typedef __attribute__((ext_vector_type(16))) __bf16          v16bf;
typedef __attribute__((ext_vector_type(16))) unsigned short  v16u;
typedef __attribute__((ext_vector_type(8)))  float           v8f;

// f32 -> bf16 RNE via hardware conversion (fptrunc); backend picks native cvt
__device__ __forceinline__ v16bf cvt16(float4 a0, float4 a1, float4 a2, float4 a3) {
    v16bf r;
    r[0]  = (__bf16)a0.x; r[1]  = (__bf16)a0.y; r[2]  = (__bf16)a0.z; r[3]  = (__bf16)a0.w;
    r[4]  = (__bf16)a1.x; r[5]  = (__bf16)a1.y; r[6]  = (__bf16)a1.z; r[7]  = (__bf16)a1.w;
    r[8]  = (__bf16)a2.x; r[9]  = (__bf16)a2.y; r[10] = (__bf16)a2.z; r[11] = (__bf16)a2.w;
    r[12] = (__bf16)a3.x; r[13] = (__bf16)a3.y; r[14] = (__bf16)a3.z; r[15] = (__bf16)a3.w;
    return r;
}

__device__ __forceinline__ float sigmoidf_(float x) {
    return 1.0f / (1.0f + __expf(-x));
}

// ---------------------------------------------------------------------------
// Kernel 1: convert V_w / U_w (f32 [32][1024]) into bf16 B-fragments in ws,
// laid out per the CDNA5 16-bit B-matrix VGPR layout:
//   fragment id  = (tile*32 + kb)*32 + lane,  16 bf16 per fragment
//   tile 0: V cols 0-15 | 1: V cols 16-31 | 2: U cols 0-15 | 3: U cols 16-31
//   lane L holds column n = L&15; h = L>>4; element i<8 holds K = kb*32 + 8h + i,
//   element i>=8 holds K = kb*32 + 16 + 8h + (i-8)
// Also zeros d_out (pool kernel accumulates with atomics).
// ---------------------------------------------------------------------------
__global__ void __launch_bounds__(256) prep_kernel(const float* __restrict__ Vw,
                                                   const float* __restrict__ Uw,
                                                   unsigned short* __restrict__ wsB,
                                                   float* __restrict__ outZero) {
    int idx = blockIdx.x * blockDim.x + threadIdx.x;
    if (idx < NUM_IN) outZero[idx] = 0.0f;
    if (idx >= 4 * 32 * 32 * 16) return;
    int i    =  idx        & 15;
    int lane = (idx >> 4)  & 31;
    int kb   = (idx >> 9)  & 31;
    int tile =  idx >> 14;
    int n = ((tile & 1) << 4) + (lane & 15);
    int k = (kb << 5) + ((lane >> 4) << 3) + (i & 7) + ((i >> 3) << 4);
    const float* W = (tile < 2) ? Vw : Uw;
    float f = W[n * NUM_IN + k];
    unsigned short h = __builtin_bit_cast(unsigned short,
                        (__bf16)f);  // hardware RNE conversion
    wsB[idx] = h;
}

// ---------------------------------------------------------------------------
// Kernel 2: per-row attention logits via v_wmma_f32_16x16x32_bf16.
// 256 threads = 8 waves; each wave owns TWO 16-row tiles (32 rows).
// All bf16 B-fragments (128 KB) are staged in dynamic LDS once per block and
// served by ds_load_b128; the x-stream (global) pipelines independently.
// ---------------------------------------------------------------------------
__global__ void __launch_bounds__(256) logits_kernel(const float* __restrict__ x,
                                                     const unsigned short* __restrict__ wsB,
                                                     const float* __restrict__ Vb,
                                                     const float* __restrict__ Ub,
                                                     const float* __restrict__ ww,
                                                     const float* __restrict__ wb,
                                                     float* __restrict__ logits,
                                                     int nTiles) {
    extern __shared__ unsigned short ldsB[];   // 65536 entries = 128 KB

    const int tid = threadIdx.x;
    // cooperative stage of all B-fragments into LDS (8192 uint4 = 128 KB)
    {
        const uint4* src = (const uint4*)wsB;
        uint4*       dst = (uint4*)ldsB;
        #pragma unroll 4
        for (int i = tid; i < 8192; i += 256) dst[i] = src[i];
    }
    __syncthreads();

    const int lane = tid & 31;
    const int wave = tid >> 5;
    const int nPairs = (nTiles + 1) >> 1;
    const int pt = blockIdx.x * 8 + wave;      // tile-pair index
    if (pt >= nPairs) return;

    const int rt0 = pt * 2;
    const int rt1v = rt0 + 1;
    const int rt1 = (rt1v < nTiles) ? rt1v : rt0;   // clamp (branch-free loop)

    const int grp = lane >> 4;          // half-wave group
    const int n   = lane & 15;          // column index / row-in-tile for A
    const int klo = grp << 3;           // A-layout K shift per half-wave

    const float* xrow0 = x + (size_t)(rt0 * 16 + n) * NUM_IN;
    const float* xrow1 = x + (size_t)(rt1 * 16 + n) * NUM_IN;
    const v16u*  B     = (const v16u*)ldsB;

    v8f c0V0 = {}, c0V1 = {}, c0U0 = {}, c0U1 = {};
    v8f c1V0 = {}, c1V1 = {}, c1U0 = {}, c1U1 = {};

    #pragma unroll 4
    for (int kb = 0; kb < 32; ++kb) {
        const int k0 = kb << 5;
        if ((kb & 3) == 0 && kb + 4 < 32) {
            __builtin_prefetch(xrow0 + k0 + 128, 0, 1);   // global_prefetch_b8
            __builtin_prefetch(xrow1 + k0 + 128, 0, 1);
        }

        // A fragments (16-bit A layout): elems 0-7 <- K=k0+klo+i,
        // elems 8-15 <- K=k0+16+klo+i
        const float4* p0 = (const float4*)(xrow0 + k0 + klo);
        const float4* q0 = (const float4*)(xrow0 + k0 + 16 + klo);
        const float4* p1 = (const float4*)(xrow1 + k0 + klo);
        const float4* q1 = (const float4*)(xrow1 + k0 + 16 + klo);
        v16bf a0 = cvt16(p0[0], p0[1], q0[0], q0[1]);
        v16bf a1 = cvt16(p1[0], p1[1], q1[0], q1[1]);

        v16bf b0 = __builtin_bit_cast(v16bf, B[              (kb << 5) + lane]);
        v16bf b1 = __builtin_bit_cast(v16bf, B[1 * 32 * 32 + (kb << 5) + lane]);
        v16bf b2 = __builtin_bit_cast(v16bf, B[2 * 32 * 32 + (kb << 5) + lane]);
        v16bf b3 = __builtin_bit_cast(v16bf, B[3 * 32 * 32 + (kb << 5) + lane]);

        c0V0 = __builtin_amdgcn_wmma_f32_16x16x32_bf16(false, a0, false, b0, (short)0, c0V0, false, false);
        c1V0 = __builtin_amdgcn_wmma_f32_16x16x32_bf16(false, a1, false, b0, (short)0, c1V0, false, false);
        c0V1 = __builtin_amdgcn_wmma_f32_16x16x32_bf16(false, a0, false, b1, (short)0, c0V1, false, false);
        c1V1 = __builtin_amdgcn_wmma_f32_16x16x32_bf16(false, a1, false, b1, (short)0, c1V1, false, false);
        c0U0 = __builtin_amdgcn_wmma_f32_16x16x32_bf16(false, a0, false, b2, (short)0, c0U0, false, false);
        c1U0 = __builtin_amdgcn_wmma_f32_16x16x32_bf16(false, a1, false, b2, (short)0, c1U0, false, false);
        c0U1 = __builtin_amdgcn_wmma_f32_16x16x32_bf16(false, a0, false, b3, (short)0, c0U1, false, false);
        c1U1 = __builtin_amdgcn_wmma_f32_16x16x32_bf16(false, a1, false, b3, (short)0, c1U1, false, false);
    }

    // C/D layout: lane L holds column n=L&15; VGPR j holds row m = j + 8*grp.
    const float vb0 = Vb[n], vb1 = Vb[n + 16];
    const float ub0 = Ub[n], ub1 = Ub[n + 16];
    const float w0  = ww[n], w1  = ww[n + 16];
    const float bias = wb[0];

    #pragma unroll
    for (int j = 0; j < 8; ++j) {
        float t0 = tanhf(c0V0[j] + vb0) * sigmoidf_(c0U0[j] + ub0) * w0
                 + tanhf(c0V1[j] + vb1) * sigmoidf_(c0U1[j] + ub1) * w1;
        float t1 = tanhf(c1V0[j] + vb0) * sigmoidf_(c1U0[j] + ub0) * w0
                 + tanhf(c1V1[j] + vb1) * sigmoidf_(c1U1[j] + ub1) * w1;
        // reduce across the 16 columns (stays within each half-wave)
        t0 += __shfl_xor(t0, 1);  t1 += __shfl_xor(t1, 1);
        t0 += __shfl_xor(t0, 2);  t1 += __shfl_xor(t1, 2);
        t0 += __shfl_xor(t0, 4);  t1 += __shfl_xor(t1, 4);
        t0 += __shfl_xor(t0, 8);  t1 += __shfl_xor(t1, 8);
        if (n == j) {
            logits[rt0 * 16 + (grp << 3) + j] = t0 + bias;
            if (rt1v < nTiles)
                logits[rt1v * 16 + (grp << 3) + j] = t1 + bias;
        }
    }
}

// ---------------------------------------------------------------------------
// Kernel 3: softmax statistics (global max, 1/sum of exp) — one block.
// ---------------------------------------------------------------------------
__global__ void __launch_bounds__(1024) stats_kernel(const float* __restrict__ logits,
                                                     int N, float* __restrict__ stats) {
    __shared__ float red[1024];
    const int tid = threadIdx.x;

    float mx = -3.0e38f;
    for (int i = tid; i < N; i += 1024) mx = fmaxf(mx, logits[i]);
    red[tid] = mx; __syncthreads();
    for (int s = 512; s > 0; s >>= 1) {
        if (tid < s) red[tid] = fmaxf(red[tid], red[tid + s]);
        __syncthreads();
    }
    const float M = red[0];
    __syncthreads();

    float sm = 0.0f;
    for (int i = tid; i < N; i += 1024) sm += __expf(logits[i] - M);
    red[tid] = sm; __syncthreads();
    for (int s = 512; s > 0; s >>= 1) {
        if (tid < s) red[tid] += red[tid + s];
        __syncthreads();
    }
    if (tid == 0) { stats[0] = M; stats[1] = 1.0f / red[0]; }
}

// ---------------------------------------------------------------------------
// Kernel 4: m = softmax(a) @ x. 256-row chunk per block; thread t owns
// features 4t..4t+3; atomic f32 accumulation into zeroed d_out.
// ---------------------------------------------------------------------------
__global__ void __launch_bounds__(256) pool_kernel(const float* __restrict__ x,
                                                   const float* __restrict__ logits,
                                                   const float* __restrict__ stats,
                                                   float* __restrict__ out, int N) {
    const int tid = threadIdx.x;
    const float M = stats[0], invS = stats[1];

    float4 acc = {0.0f, 0.0f, 0.0f, 0.0f};
    const int r0 = blockIdx.x * 256;
    const int r1 = (r0 + 256 < N) ? (r0 + 256) : N;
    for (int r = r0; r < r1; ++r) {
        const float w = __expf(logits[r] - M) * invS;
        const float4 v = ((const float4*)(x + (size_t)r * NUM_IN))[tid];
        acc.x += w * v.x; acc.y += w * v.y; acc.z += w * v.z; acc.w += w * v.w;
    }
    atomicAdd(&out[tid * 4 + 0], acc.x);
    atomicAdd(&out[tid * 4 + 1], acc.y);
    atomicAdd(&out[tid * 4 + 2], acc.z);
    atomicAdd(&out[tid * 4 + 3], acc.w);
}

// ---------------------------------------------------------------------------
extern "C" void kernel_launch(void* const* d_in, const int* in_sizes, int n_in,
                              void* d_out, int out_size, void* d_ws, size_t ws_size,
                              hipStream_t stream) {
    const float* x  = (const float*)d_in[0];
    const float* Vw = (const float*)d_in[1];
    const float* Vb = (const float*)d_in[2];
    const float* Uw = (const float*)d_in[3];
    const float* Ub = (const float*)d_in[4];
    const float* ww = (const float*)d_in[5];
    const float* wb = (const float*)d_in[6];
    float* out = (float*)d_out;

    const int N = in_sizes[0] / NUM_IN;          // 200000 (divisible by 32)
    const int nTiles = N / 16;
    const int Nfull  = nTiles * 16;
    const int nPairs = (nTiles + 1) / 2;

    // ws layout: [0,128KB) bf16 B-fragments | stats (2 f32) | logits (N f32)
    unsigned short* wsB = (unsigned short*)d_ws;
    float* wsf    = (float*)d_ws;
    float* stats  = wsf + 32768;
    float* logits = wsf + 32768 + 32;

    prep_kernel<<<(4 * 32 * 32 * 16 + 255) / 256, 256, 0, stream>>>(Vw, Uw, wsB, out);
    logits_kernel<<<(nPairs + 7) / 8, 256, 131072, stream>>>(x, wsB, Vb, Ub, ww, wb,
                                                             logits, nTiles);
    stats_kernel<<<1, 1024, 0, stream>>>(logits, Nfull, stats);
    pool_kernel<<<(Nfull + 255) / 256, 256, 0, stream>>>(x, logits, stats, out, Nfull);
}
